// InnerFuncAttn_19344532702114
// MI455X (gfx1250) — matrix operand order repaired and checked by reading, not compiled, not run on hardware
//
#include <hip/hip_runtime.h>
#include <hip/hip_bf16.h>

typedef __attribute__((ext_vector_type(16))) _Float16 v16h;
typedef __attribute__((ext_vector_type(8)))  _Float16 v8h;
typedef __attribute__((ext_vector_type(8)))  float    v8f;
typedef __attribute__((ext_vector_type(4)))  unsigned u32x4;
typedef __attribute__((ext_vector_type(8)))  unsigned u32x8;

#define S_LEN 2048
#define DMODEL 1024
#define NHEADS 16
#define HDIM 64
#define NVALS 8192
#define DRETR 128

__device__ __forceinline__ v16h cat8(v8h lo, v8h hi) {
    return __builtin_shufflevector(lo, hi, 0,1,2,3,4,5,6,7,8,9,10,11,12,13,14,15);
}

// A-operand fragment (16x32 f16, CDNA5 layout): elements 0..7 = K[kg..kg+7],
// elements 8..15 = K[16+kg..16+kg+7], kg = (lane>=16) ? 8 : 0.
__device__ __forceinline__ v16h load_afrag(const _Float16* row, int kg) {
    v8h lo = *(const v8h*)(row + kg);
    v8h hi = *(const v8h*)(row + 16 + kg);
    return cat8(lo, hi);
}

// gfx1250 async global->LDS copy, 16 bytes per lane. LDS offset = low 32 bits
// of the flat shared-memory address (ISA 10.2: LDS aperture truncation).
__device__ __forceinline__ void async_copy16(const void* g, void* lds) {
    unsigned lds_off = (unsigned)(size_t)lds;
    unsigned long long ga = (unsigned long long)(size_t)g;
    asm volatile("global_load_async_to_lds_b128 %0, %1, off"
                 :: "v"(lds_off), "v"(ga) : "memory");
}
__device__ __forceinline__ void async_wait0() {
    asm volatile("s_wait_asynccnt 0x0" ::: "memory");
}
__device__ __forceinline__ void async_wait1() {   // allow 1 newer set in flight
    asm volatile("s_wait_asynccnt 0x1" ::: "memory");
}

// Tensor Data Mover: load a 2-D f16 tile rows x cols (elements) with row pitch
// `stride` (elements) from global into LDS. One issue per calling wave;
// completion tracked by TENSORcnt. D# per CDNA5 ISA ch.8 (2 SGPR groups).
__device__ __forceinline__ void tdm_load_tile_f16(
    const void* g, void* lds, unsigned rows, unsigned cols, unsigned stride)
{
    unsigned long long ga = (unsigned long long)(size_t)g;
    u32x4 g0;
    g0[0] = 1u;                                      // count=1, user mode
    g0[1] = (unsigned)(size_t)lds;                   // lds_addr (bytes)
    g0[2] = (unsigned)ga;                            // global_addr[31:0]
    g0[3] = (unsigned)((ga >> 32) & 0x1FFFFFFu) | (2u << 30);  // addr[56:32], type=2
    u32x8 g1;
    g1[0] = 1u << 16;          // workgroup_mask=0, data_size=1 (2 bytes)
    g1[1] = cols << 16;        // tensor_dim0[15:0]
    g1[2] = rows << 16;        // tensor_dim0[31:16]=0, tensor_dim1[15:0]
    g1[3] = cols << 16;        // tensor_dim1[31:16]=0, tile_dim0
    g1[4] = rows;              // tile_dim1, tile_dim2=0
    g1[5] = stride;            // tensor_dim0_stride[31:0]
    g1[6] = 0u;                // stride[47:32]=0, tensor_dim1_stride lo
    g1[7] = 0u;
    asm volatile("tensor_load_to_lds %0, %1" :: "s"(g0), "s"(g1) : "memory");
}

// f32 -> f16 conversion pass (4 elements / thread)
__global__ __launch_bounds__(256) void cvt_f16_kernel(
    const float* __restrict__ src, _Float16* __restrict__ dst, int n)
{
    int i = (blockIdx.x * 256 + threadIdx.x) * 4;
    if (i < n) {
        float4 f = *(const float4*)(src + i);
        dst[i + 0] = (_Float16)f.x;
        dst[i + 1] = (_Float16)f.y;
        dst[i + 2] = (_Float16)f.z;
        dst[i + 3] = (_Float16)f.w;
    }
}

// ---------------------------------------------------------------------------
// Tiled GEMM: C[M,N] = A[M,K] @ W[K,N] + bias. A,W f16; output f32 or f16.
// Block tile 64x128, 8 waves (2x4), each wave owns a 32x32 sub-tile.
// Double-buffered pipeline: A tile via per-lane async copies (ASYNCcnt),
// B tile via one TDM tensor_load_to_lds from wave 0 (TENSORcnt); next tile's
// transfers fly while current tile is in the WMMA units.
// ---------------------------------------------------------------------------
__global__ __launch_bounds__(256) void gemm_wmma_kernel(
    const _Float16* __restrict__ A, const _Float16* __restrict__ W,
    const float* __restrict__ bias, void* __restrict__ Cout,
    int M, int N, int K, int outHalf)
{
    __shared__ __align__(32) _Float16 sA[2][64][32];
    __shared__ __align__(32) _Float16 sB[2][32][128];

    const int t  = threadIdx.x;
    const int l  = t & 31;
    const int w  = t >> 5;
    const int wm = w >> 2;        // 0..1
    const int wn = w & 3;         // 0..3
    const int hi = l >> 4;
    const int ln = l & 15;
    const int m0 = blockIdx.y * 64;
    const int n0 = blockIdx.x * 128;
    const int kg = hi * 8;

    const int ar = (t * 8) >> 5,  ac = (t * 8) & 31;    // A staging: 16B/thread
    const int br = (t * 16) >> 7, bc = (t * 16) & 127;  // B prefetch coords

    v8f acc[2][2] = {};

    // prologue: stage k-chunk 0 into buffer 0
    async_copy16(A + (size_t)(m0 + ar) * K + ac, &sA[0][ar][ac]);
    if (w == 0)
        tdm_load_tile_f16(W + n0, &sB[0][0][0], 32, 128, (unsigned)N);

    int buf = 0;
    for (int kc = 0; kc < K; kc += 32) {
        const bool more = (kc + 32) < K;
        if (more) {
            async_copy16(A + (size_t)(m0 + ar) * K + kc + 32 + ac,
                         &sA[buf ^ 1][ar][ac]);
            if (w == 0)
                tdm_load_tile_f16(W + (size_t)(kc + 32) * N + n0,
                                  &sB[buf ^ 1][0][0], 32, 128, (unsigned)N);
        }
        if (kc + 64 < K) {   // warm L2 two tiles ahead
            __builtin_prefetch(A + (size_t)(m0 + ar) * K + kc + 64 + ac, 0, 1);
            __builtin_prefetch(W + (size_t)(kc + 64 + br) * N + n0 + bc, 0, 1);
        }
        // wait for current buffer only (1 newer transfer set may stay in flight)
        if (more) async_wait1(); else async_wait0();
        if (w == 0) {
            if (more) __builtin_amdgcn_s_wait_tensorcnt(1);
            else      __builtin_amdgcn_s_wait_tensorcnt(0);
        }
        __syncthreads();

        for (int i = 0; i < 2; ++i) {
            v16h a = load_afrag(&sA[buf][wm * 32 + i * 16 + ln][0], kg);
            for (int j = 0; j < 2; ++j) {
                v16h bfrag;
                for (int s = 0; s < 16; ++s)
                    bfrag[s] = sB[buf][s < 8 ? kg + s : 8 + kg + s][wn * 32 + j * 16 + ln];
                acc[i][j] = __builtin_amdgcn_wmma_f32_16x16x32_f16(
                    false, a, false, bfrag, (short)0, acc[i][j], false, false);
            }
        }
        __syncthreads();
        buf ^= 1;
    }

    if (outHalf) {
        _Float16* C = (_Float16*)Cout;
        for (int i = 0; i < 2; ++i)
            for (int j = 0; j < 2; ++j) {
                int n = n0 + wn * 32 + j * 16 + ln;
                float bv = bias[n];
                for (int v = 0; v < 8; ++v) {
                    int m = m0 + wm * 32 + i * 16 + v + 8 * hi;
                    C[(size_t)m * N + n] = (_Float16)(acc[i][j][v] + bv);
                }
            }
    } else {
        float* C = (float*)Cout;
        for (int i = 0; i < 2; ++i)
            for (int j = 0; j < 2; ++j) {
                int n = n0 + wn * 32 + j * 16 + ln;
                float bv = bias[n];
                for (int v = 0; v < 8; ++v) {
                    int m = m0 + wm * 32 + i * 16 + v + 8 * hi;
                    C[(size_t)m * N + n] = acc[i][j][v] + bv;
                }
            }
    }
}

// ---------------------------------------------------------------------------
// Top-1 retrieval: one workgroup per token row; argmax over 8192 keys with
// first-max tie-break (matches jnp.argmax).
// ---------------------------------------------------------------------------
__global__ __launch_bounds__(256) void top1_kernel(
    const float* __restrict__ VQ, const float* __restrict__ VKeys,
    int* __restrict__ idxOut)
{
    __shared__ float srow[DRETR];
    __shared__ float sval[256];
    __shared__ int   sidx[256];

    const int row = blockIdx.x;
    const int t = threadIdx.x;
    if (t < DRETR) srow[t] = VQ[(size_t)row * DRETR + t];
    __syncthreads();

    float best = -3.4e38f;
    int bi = 0;
    for (int n = t; n < NVALS; n += 256) {
        const float* kp = VKeys + (size_t)n * DRETR;
        float dot = 0.0f;
        for (int r = 0; r < DRETR; r += 4) {
            float4 kv = *(const float4*)(kp + r);
            dot += srow[r] * kv.x + srow[r + 1] * kv.y +
                   srow[r + 2] * kv.z + srow[r + 3] * kv.w;
        }
        if (dot > best) { best = dot; bi = n; }
    }
    sval[t] = best; sidx[t] = bi;
    __syncthreads();
    for (int s = 128; s > 0; s >>= 1) {
        if (t < s) {
            float ov = sval[t + s]; int oi = sidx[t + s];
            if (ov > sval[t] || (ov == sval[t] && oi < sidx[t])) {
                sval[t] = ov; sidx[t] = oi;
            }
        }
        __syncthreads();
    }
    if (t == 0) idxOut[row] = sidx[0];
}

// v = hs * v_embed[idx]  -> f16 (elementwise over [B*S, D])
__global__ __launch_bounds__(256) void gather_mul_kernel(
    const float* __restrict__ hs, const float* __restrict__ vemb,
    const int* __restrict__ idx, _Float16* __restrict__ V, int total)
{
    int i = blockIdx.x * 256 + threadIdx.x;
    if (i < total) {
        int row = i >> 10;          // / DMODEL
        int d = i & (DMODEL - 1);
        V[i] = (_Float16)(hs[i] * vemb[(size_t)idx[row] * DMODEL + d]);
    }
}

// ---------------------------------------------------------------------------
// Flash attention, causal + per-head dynamic mask. One block per
// (b, h, 128-row q-block); each wave owns a 16-row q-tile; waves share staged
// K/V tiles of 32 keys. Q/K/V are f16; online softmax in f32; f16 output.
// ---------------------------------------------------------------------------
__global__ __launch_bounds__(256) void flash_attn_kernel(
    const _Float16* __restrict__ Q, const _Float16* __restrict__ Km,
    const _Float16* __restrict__ V, const float* __restrict__ amask,
    const float* __restrict__ dmask, _Float16* __restrict__ O)
{
    __shared__ __align__(32) _Float16 sK[HDIM][32];   // [d][key] (transposed)
    __shared__ __align__(32) _Float16 sV[32][HDIM];   // [key][d]
    __shared__ __align__(32) _Float16 sP[8][16][32];  // per-wave P tile
    __shared__ float sAM[32];

    const int bid = blockIdx.x;
    const int qb = bid & 15;
    const int h  = (bid >> 4) & (NHEADS - 1);
    const int b  = bid >> 8;
    const int t = threadIdx.x, l = t & 31, w = t >> 5;
    const int hi = l >> 4, ln = l & 15;
    const int q0 = qb * 128;
    const int qw = q0 + w * 16;       // this wave's first q row
    const int kg = hi * 8;

    // Q fragments straight from f16 global memory in A layout (no conversion)
    v16h qf[2];
    {
        const _Float16* qg = Q + ((size_t)(b * S_LEN + qw + ln) * DMODEL) + h * HDIM;
        for (int c = 0; c < 2; ++c)
            qf[c] = cat8(*(const v8h*)(qg + c * 32 + kg),
                         *(const v8h*)(qg + c * 32 + 16 + kg));
    }

    v8f oacc[4] = {};
    float mrun[8], lrun[8];
    for (int v = 0; v < 8; ++v) { mrun[v] = -3.0e38f; lrun[v] = 0.0f; }

    const int jend = q0 + 128;  // causal bound for this q-block
    for (int j0 = 0; j0 < jend; j0 += 32) {
        // ---- staging: K transposed via registers, V via async copy ----
        {
            int key = t >> 3, ds = (t & 7) * 8;
            const _Float16* kgp = Km + ((size_t)(b * S_LEN + j0 + key) * DMODEL) + h * HDIM + ds;
            const _Float16* vgp = V  + ((size_t)(b * S_LEN + j0 + key) * DMODEL) + h * HDIM + ds;
            async_copy16(vgp, &sV[key][ds]);
            v8h kv = *(const v8h*)kgp;
            for (int e = 0; e < 8; ++e) sK[ds + e][key] = kv[e];
            if (t < 32) {
                int j = j0 + t;
                sAM[t] = amask[b * S_LEN + j] * dmask[h * S_LEN + j];
            }
        }
        async_wait0();
        __syncthreads();

        if (j0 <= qw + 15) {   // wave-uniform causal tile skip
            // ---- S[16 q x 32 keys] = Q @ K^T (2 n-subtiles) ----
            v8f sacc[2] = {};
            for (int sub = 0; sub < 2; ++sub)
                for (int c = 0; c < 2; ++c) {
                    v16h bf;
                    for (int s = 0; s < 16; ++s)
                        bf[s] = sK[c * 32 + (s < 8 ? kg + s : 8 + kg + s)][sub * 16 + ln];
                    sacc[sub] = __builtin_amdgcn_wmma_f32_16x16x32_f16(
                        false, qf[c], false, bf, (short)0, sacc[sub], false, false);
                }
            // ---- mask + scale ----
            float tmax[8];
            for (int v = 0; v < 8; ++v) {
                int irow = qw + v + 8 * hi;
                int jA = j0 + ln, jB = j0 + 16 + ln;
                float s0 = sacc[0][v] * 0.125f;
                float s1 = sacc[1][v] * 0.125f;
                bool okA = (jA <= irow) && (sAM[ln] != 0.0f);
                bool okB = (jB <= irow) && (sAM[16 + ln] != 0.0f);
                s0 = okA ? s0 : -3.0e38f;
                s1 = okB ? s1 : -3.0e38f;
                sacc[0][v] = s0; sacc[1][v] = s1;
                tmax[v] = fmaxf(s0, s1);
            }
            for (int off = 1; off < 16; off <<= 1)
                for (int v = 0; v < 8; ++v)
                    tmax[v] = fmaxf(tmax[v], __shfl_xor(tmax[v], off, 32));
            float scale[8], rsum[8];
            for (int v = 0; v < 8; ++v) {
                float mt = fmaxf(mrun[v], tmax[v]);
                scale[v] = __expf(mrun[v] - mt);
                mrun[v] = mt;
                float p0 = __expf(sacc[0][v] - mt);
                float p1 = __expf(sacc[1][v] - mt);
                sacc[0][v] = p0; sacc[1][v] = p1;
                rsum[v] = p0 + p1;
            }
            for (int off = 1; off < 16; off <<= 1)
                for (int v = 0; v < 8; ++v)
                    rsum[v] += __shfl_xor(rsum[v], off, 32);
            for (int v = 0; v < 8; ++v) lrun[v] = lrun[v] * scale[v] + rsum[v];
            for (int tt = 0; tt < 4; ++tt)
                for (int v = 0; v < 8; ++v)
                    oacc[tt][v] *= scale[v];

            // ---- P through LDS (D-layout -> A-layout transpose) ----
            for (int sub = 0; sub < 2; ++sub)
                for (int v = 0; v < 8; ++v)
                    sP[w][v + 8 * hi][sub * 16 + ln] = (_Float16)sacc[sub][v];
            asm volatile("s_wait_dscnt 0x0" ::: "memory");

            // ---- O += P @ V (4 d-subtiles) ----
            v16h pf = load_afrag(&sP[w][ln][0], kg);
            for (int tt = 0; tt < 4; ++tt) {
                v16h bf;
                for (int s = 0; s < 16; ++s)
                    bf[s] = sV[s < 8 ? kg + s : 8 + kg + s][tt * 16 + ln];
                oacc[tt] = __builtin_amdgcn_wmma_f32_16x16x32_f16(
                    false, pf, false, bf, (short)0, oacc[tt], false, false);
            }
        }
        __syncthreads();
    }

    // normalize + write f16 [b, s, h*64+d] (== transpose+reshape)
    for (int tt = 0; tt < 4; ++tt)
        for (int v = 0; v < 8; ++v) {
            int irow = qw + v + 8 * hi;
            float inv = 1.0f / lrun[v];
            O[((size_t)(b * S_LEN + irow) * DMODEL) + h * HDIM + tt * 16 + ln] =
                (_Float16)(oacc[tt][v] * inv);
        }
}

// ---------------------------------------------------------------------------
extern "C" void kernel_launch(void* const* d_in, const int* in_sizes, int n_in,
                              void* d_out, int out_size, void* d_ws, size_t ws_size,
                              hipStream_t stream) {
    (void)in_sizes; (void)n_in; (void)out_size; (void)ws_size;
    const float* hs    = (const float*)d_in[0];
    const float* amask = (const float*)d_in[1];
    const float* Wq    = (const float*)d_in[2];
    const float* bq    = (const float*)d_in[3];
    const float* Wk    = (const float*)d_in[4];
    const float* bk    = (const float*)d_in[5];
    const float* dmask = (const float*)d_in[6];
    const float* Wvq   = (const float*)d_in[7];
    const float* bvq   = (const float*)d_in[8];
    const float* vkeys = (const float*)d_in[9];
    const float* vemb  = (const float*)d_in[10];
    const float* Wo    = (const float*)d_in[11];
    const float* bo    = (const float*)d_in[12];
    float* out = (float*)d_out;

    const int M = 2 * S_LEN;  // 4096 token rows
    char* ws = (char*)d_ws;
    _Float16* hsH  = (_Float16*)ws; ws += (size_t)M * DMODEL * 2;
    _Float16* WqH  = (_Float16*)ws; ws += (size_t)DMODEL * DMODEL * 2;
    _Float16* WkH  = (_Float16*)ws; ws += (size_t)DMODEL * DMODEL * 2;
    _Float16* WvqH = (_Float16*)ws; ws += (size_t)DMODEL * DRETR * 2;
    _Float16* WoH  = (_Float16*)ws; ws += (size_t)DMODEL * DMODEL * 2;
    _Float16* QH   = (_Float16*)ws; ws += (size_t)M * DMODEL * 2;
    _Float16* KH   = (_Float16*)ws; ws += (size_t)M * DMODEL * 2;
    _Float16* VH   = (_Float16*)ws; ws += (size_t)M * DMODEL * 2;
    _Float16* AOH  = (_Float16*)ws; ws += (size_t)M * DMODEL * 2;
    float*    VQb  = (float*)ws;    ws += (size_t)M * DRETR * sizeof(float);
    int*      idxb = (int*)ws;

    dim3 blk(256);
    int nHS = M * DMODEL, nW = DMODEL * DMODEL, nWv = DMODEL * DRETR;
    cvt_f16_kernel<<<dim3((nHS / 4 + 255) / 256), blk, 0, stream>>>(hs, hsH, nHS);
    cvt_f16_kernel<<<dim3((nW / 4 + 255) / 256), blk, 0, stream>>>(Wq, WqH, nW);
    cvt_f16_kernel<<<dim3((nW / 4 + 255) / 256), blk, 0, stream>>>(Wk, WkH, nW);
    cvt_f16_kernel<<<dim3((nWv / 4 + 255) / 256), blk, 0, stream>>>(Wvq, WvqH, nWv);
    cvt_f16_kernel<<<dim3((nW / 4 + 255) / 256), blk, 0, stream>>>(Wo, WoH, nW);

    gemm_wmma_kernel<<<dim3(DMODEL / 128, M / 64), blk, 0, stream>>>(
        hsH, WqH, bq, QH, M, DMODEL, DMODEL, 1);
    gemm_wmma_kernel<<<dim3(DMODEL / 128, M / 64), blk, 0, stream>>>(
        hsH, WkH, bk, KH, M, DMODEL, DMODEL, 1);
    gemm_wmma_kernel<<<dim3(DRETR / 128, M / 64), blk, 0, stream>>>(
        hsH, WvqH, bvq, VQb, M, DRETR, DMODEL, 0);
    top1_kernel<<<dim3(M), blk, 0, stream>>>(VQb, vkeys, idxb);
    gather_mul_kernel<<<dim3((M * DMODEL + 255) / 256), blk, 0, stream>>>(
        hs, vemb, idxb, VH, M * DMODEL);
    flash_attn_kernel<<<dim3(2 * NHEADS * (S_LEN / 128)), blk, 0, stream>>>(
        QH, KH, VH, amask, dmask, AOH);
    gemm_wmma_kernel<<<dim3(DMODEL / 128, M / 64), blk, 0, stream>>>(
        AOH, WoH, bo, out, M, DMODEL, DMODEL, 0);
}
